// Program_72902774882565
// MI455X (gfx1250) — matrix operand order repaired and checked by reading, not compiled, not run on hardware
//
#include <hip/hip_runtime.h>

typedef __attribute__((ext_vector_type(16))) _Float16 v16h;
typedef __attribute__((ext_vector_type(8)))  _Float16 v8h;
typedef __attribute__((ext_vector_type(8)))  float    v8f;

namespace {
constexpr int   kSteps       = 10;
constexpr float kDT          = 0.02f;
constexpr float kGrav        = 9.8f;
constexpr float kTotalMass   = 1.1f;
constexpr float kPoleMassLen = 0.05f;
constexpr float kMassPole    = 0.1f;
}

// Per-wave LDS regions are private -> in-wave DS-counter wait is enough.
__device__ __forceinline__ void lds_fence() {
  asm volatile("s_wait_dscnt 0x0" ::: "memory");
}

__device__ __forceinline__ v8f wmma16(v16h a, v16h b, v8f c) {
  // (neg_a, A, neg_b, B, c_mod, C, reuse_a, reuse_b)
  return __builtin_amdgcn_wmma_f32_16x16x32_f16(false, a, false, b, (short)0, c,
                                                false, false);
}

// A-fragment (16x32 f16): lane L holds row M=L%16; elements 0..15 = K0..K0+15,
// with K0 = 32*chunk + 16*(L>>4).
__device__ __forceinline__ v16h load_a_frag(const _Float16* row, int k0) {
  v8h lo = *(const v8h*)(row + k0);
  v8h h8 = *(const v8h*)(row + k0 + 8);
  return __builtin_shufflevector(lo, h8, 0, 1, 2, 3, 4, 5, 6, 7,
                                 8, 9, 10, 11, 12, 13, 14, 15);
}

__global__ __launch_bounds__(256) void cartpole_policy_rollout(
    const float* __restrict__ inp,
    const float* __restrict__ w1p, const float* __restrict__ b1p,
    const float* __restrict__ w2p, const float* __restrict__ b2p,
    const float* __restrict__ w3p, const float* __restrict__ b3p,
    float* __restrict__ out, int B) {
  // Row stride 80 f16 (160B): keeps 16B alignment for ds_load_b128 and
  // spreads the per-row LDS banks.
  __shared__ __align__(16) _Float16 hb1[8][16][80];
  __shared__ __align__(16) _Float16 hb2[8][16][80];
  __shared__ float lgs[8][16];

  const int lane = threadIdx.x & 31;
  const int wv   = threadIdx.x >> 5;   // wave in block (0..7)
  const int col  = lane & 15;          // N-column / env-in-tile
  const int hi   = lane >> 4;          // half-wave (0/1)
  const int env  = blockIdx.x * 128 + wv * 16 + col;
  const bool owner = (hi == 0) && (env < B);

  // ---------------- resident weight fragments (loaded once) ---------------
  // B-fragment (K x 16 f16): lane L holds column N=L%16; element e = K row
  // (16*hi + e) within the 32-row chunk.

  // Layer 1: w1 is 4x64; K rows 4..31 are zero padding.
  v16h B1f[4];
#pragma unroll
  for (int n = 0; n < 4; ++n) {
    v16h f = {};
    const int c = n * 16 + col;
    if (hi == 0) {
#pragma unroll
      for (int e = 0; e < 4; ++e) f[e] = (_Float16)w1p[e * 64 + c];
    }
    B1f[n] = f;
  }

  // Layer 2: w2 is 64x64 -> 2 K-chunks x 4 N-tiles.
  v16h B2f[2][4];
#pragma unroll
  for (int ch = 0; ch < 2; ++ch) {
#pragma unroll
    for (int n = 0; n < 4; ++n) {
      v16h f;
      const int c  = n * 16 + col;
      const int k0 = ch * 32 + hi * 16;
#pragma unroll
      for (int e = 0; e < 16; ++e) f[e] = (_Float16)w2p[(k0 + e) * 64 + c];
      B2f[ch][n] = f;
    }
  }

  // Layer 3: w3 is 64x1 -> only output column N=0 is non-zero.
  v16h B3f[2];
#pragma unroll
  for (int ch = 0; ch < 2; ++ch) {
    v16h f = {};
    if (col == 0) {
      const int k0 = ch * 32 + hi * 16;
#pragma unroll
      for (int e = 0; e < 16; ++e) f[e] = (_Float16)w3p[k0 + e];
    }
    B3f[ch] = f;
  }

  float b1v[4], b2v[4];
#pragma unroll
  for (int n = 0; n < 4; ++n) {
    b1v[n] = b1p[n * 16 + col];
    b2v[n] = b2p[n * 16 + col];
  }
  const float b3s = b3p[0];

  // ---------------- per-env state (lanes 0..15 own one env each) ----------
  float x = 0.f, xd = 0.f, th = 0.f, thd = 0.f;
  float4* outv = reinterpret_cast<float4*>(out);
  if (owner) {
    const size_t base = (size_t)env * 12;
    x   = inp[base + 1];
    xd  = inp[base + 2];
    th  = inp[base + 3];
    thd = inp[base + 4];
    outv[env] = make_float4(th, x, xd, thd);  // init_traj row (step 0)
  }

#pragma unroll 1
  for (int s = 0; s < kSteps; ++s) {
    // ---- A1: feats tile 16x32 f16 (K=0..3 valid, rest zero) -------------
    v16h a1 = {};
    a1[0] = (hi == 0) ? (_Float16)x   : (_Float16)0.f;
    a1[1] = (hi == 0) ? (_Float16)xd  : (_Float16)0.f;
    a1[2] = (hi == 0) ? (_Float16)th  : (_Float16)0.f;
    a1[3] = (hi == 0) ? (_Float16)thd : (_Float16)0.f;

    // ---- layer 1: 4 WMMAs, bias in accumulator, ReLU, f16 -> LDS --------
#pragma unroll
    for (int n = 0; n < 4; ++n) {
      v8f c0;
#pragma unroll
      for (int r = 0; r < 8; ++r) c0[r] = b1v[n];
      v8f c1 = wmma16(a1, B1f[n], c0);
#pragma unroll
      for (int r = 0; r < 8; ++r) {
        float v = fmaxf(c1[r], 0.f);
        hb1[wv][r + 8 * hi][n * 16 + col] = (_Float16)v;
      }
    }
    lds_fence();

    // ---- layer 2: rebuild A from LDS, 8 WMMAs, ReLU, f16 -> LDS ---------
    v16h a20 = load_a_frag(&hb1[wv][col][0], 0 + 16 * hi);
    v16h a21 = load_a_frag(&hb1[wv][col][0], 32 + 16 * hi);
#pragma unroll
    for (int n = 0; n < 4; ++n) {
      v8f c0;
#pragma unroll
      for (int r = 0; r < 8; ++r) c0[r] = b2v[n];
      v8f t  = wmma16(a20, B2f[0][n], c0);
      v8f c2 = wmma16(a21, B2f[1][n], t);
#pragma unroll
      for (int r = 0; r < 8; ++r) {
        float v = fmaxf(c2[r], 0.f);
        hb2[wv][r + 8 * hi][n * 16 + col] = (_Float16)v;
      }
    }
    lds_fence();

    // ---- layer 3: 2 WMMAs into column 0; logits land in lanes 0/16 ------
    v16h a30 = load_a_frag(&hb2[wv][col][0], 0 + 16 * hi);
    v16h a31 = load_a_frag(&hb2[wv][col][0], 32 + 16 * hi);
    v8f d3;
#pragma unroll
    for (int r = 0; r < 8; ++r) d3[r] = b3s;
    d3 = wmma16(a30, B3f[0], d3);
    d3 = wmma16(a31, B3f[1], d3);
    if (col == 0) {
#pragma unroll
      for (int r = 0; r < 8; ++r) lgs[wv][8 * hi + r] = d3[r];
    }
    lds_fence();

    // sigmoid(z) < 0.5  <=>  z < 0, so threshold the logit directly.
    const float logit = lgs[wv][col];
    const float force = (logit < 0.f) ? -10.f : 10.f;

    // ---- physics update (cheap VALU; upper half computes junk, unused) --
    const float sn = sinf(th);
    const float cs = cosf(th);
    const float temp = (force + sn * thd * thd * kPoleMassLen) / kTotalMass;
    const float thacc = (kGrav * sn - cs * temp) * 2.0f /
                        (4.0f / 3.0f - cs * cs * (kMassPole / kTotalMass));
    const float xacc = temp - thacc * cs * (kPoleMassLen / kTotalMass);
    const float xn   = x + xd * kDT;
    const float xdn  = xd + xacc * kDT;
    const float thn  = th + thd * kDT;
    const float thdn = thd + thacc * kDT;
    if (owner) {
      outv[(size_t)(s + 1) * B + env] = make_float4(thn, xn, xdn, thdn);
    }
    x = xn; xd = xdn; th = thn; thd = thdn;
  }
}

extern "C" void kernel_launch(void* const* d_in, const int* in_sizes, int n_in,
                              void* d_out, int out_size, void* d_ws,
                              size_t ws_size, hipStream_t stream) {
  const float* inp = (const float*)d_in[0];
  const float* w1  = (const float*)d_in[1];
  const float* b1  = (const float*)d_in[2];
  const float* w2  = (const float*)d_in[3];
  const float* b2  = (const float*)d_in[4];
  const float* w3  = (const float*)d_in[5];
  const float* b3  = (const float*)d_in[6];
  float* out = (float*)d_out;

  const int B = in_sizes[0] / 12;           // 2,097,152
  const int blocks = (B + 127) / 128;       // 128 envs per 256-thread block
  cartpole_policy_rollout<<<blocks, 256, 0, stream>>>(inp, w1, b1, w2, b2, w3,
                                                      b3, out, B);
}
